// SkeletonBlock_24180665876990
// MI455X (gfx1250) — matrix operand order repaired and checked by reading, not compiled, not run on hardware
//
#include <hip/hip_runtime.h>
#include <hip/hip_bf16.h>
#include <math.h>
#include <stdint.h>

// ---------------------------------------------------------------------------
// Problem constants (from reference)
// ---------------------------------------------------------------------------
#define D_MODEL 1024
#define N_HEADS 16
#define D_HEAD  64
#define D_FF    4096
#define NUM_FAM 64
#define DICTN   4096
#define TOP_K   8
#define BB      2
#define TT      2048
#define NTOK    (BB * TT)          // 4096 tokens
#define EPSV    1e-6f

typedef __attribute__((ext_vector_type(16))) __bf16 v16bf;
typedef __attribute__((ext_vector_type(8)))  float  v8f;

// ---------------------------------------------------------------------------
// Generic batched GEMM:  C[m,n] = sum_k A[m,k] * W[n,k]   (+ optional addend)
// A, W row-major fp32 (K contiguous). bf16 WMMA with fp32 accumulate.
//
// CDNA5 path: double-buffered LDS stages filled with GLOBAL_LOAD_ASYNC_TO_LDS
// (ASYNCcnt-tracked, no VGPR staging), f32->bf16 conversion fused into the
// fragment load (VALU co-executes with the WMMA/XDL pipe).
//
// Block tile 128x64, K-step 32, 256 threads = 8 waves (4 M x 2 N), each wave
// a 32x32 tile = 4 accumulators -> 4 v_wmma per k-step per wave.
// All call sites use M % 128 == 0, N % 64 == 0, K % 32 == 0 (no guards).
// ---------------------------------------------------------------------------
struct GemmP {
    const float* A; const float* W; float* C; const float* Add;
    int M, N, K, lda, ldw, ldc, nb1, storeT;
    long aS0, aS1, wS0, wS1, cS0, cS1;
};

#define BMT 128
#define BNT 64
#define BKT 32
#define RS  36                      // padded f32 row stride in LDS
#define STAGEF ((BMT + BNT) * RS)   // floats per stage (A tile then B tile)

__device__ __forceinline__ v16bf cvt16(const float* p0, const float* p1) {
    v16bf r;
    #pragma unroll
    for (int i = 0; i < 8; ++i) r[i] = (__bf16)p0[i];
    #pragma unroll
    for (int i = 0; i < 8; ++i) r[8 + i] = (__bf16)p1[i];
    return r;
}

__global__ __launch_bounds__(256) void gemm_bf16_wmma(GemmP p) {
    __shared__ float sm[2 * STAGEF];    // 55296 B, double-buffered A|B stages

    const int z  = blockIdx.z;
    const int b0 = z / p.nb1, b1 = z % p.nb1;
    const float* A = p.A + (long)b0 * p.aS0 + (long)b1 * p.aS1;
    const float* W = p.W + (long)b0 * p.wS0 + (long)b1 * p.wS1;
    float*       C = p.C + (long)b0 * p.cS0 + (long)b1 * p.cS1;
    const float* Add = p.Add ? p.Add + (long)b0 * p.cS0 + (long)b1 * p.cS1 : nullptr;

    const int m0 = blockIdx.y * BMT, n0 = blockIdx.x * BNT;
    const int tid  = threadIdx.x;
    const int lane = tid & 31;
    const int wid  = tid >> 5;
    const int wm   = wid & 3;      // wave M-quarter (32 rows)
    const int wn   = wid >> 2;     // wave N-half   (32 cols)
    const int half = lane >> 4;    // lane group 0/1
    const int l16  = lane & 15;

    v8f acc00 = {}, acc01 = {}, acc10 = {}, acc11 = {};

    // async-copy mapping: each lane moves 16 B; 256 threads = 4 KB per issue
    const int crow = tid >> 3;          // 0..31
    const int cc4  = (tid & 7) * 4;     // float column offset (16B granules)

    auto issue_stage = [&](int k0, int stage) {
        float* sA = sm + stage * STAGEF;
        float* sB = sA + BMT * RS;
        #pragma unroll
        for (int i = 0; i < 4; ++i) {   // A: 128 rows x 32 f32
            const int r = crow + 32 * i;
            const uint32_t l = (uint32_t)(uintptr_t)(sA + r * RS + cc4);
            const uint64_t g = (uint64_t)(uintptr_t)(A + (long)(m0 + r) * p.lda + k0 + cc4);
            asm volatile("global_load_async_to_lds_b128 %0, %1, off"
                         :: "v"(l), "v"(g) : "memory");
        }
        #pragma unroll
        for (int i = 0; i < 2; ++i) {   // B: 64 rows x 32 f32
            const int r = crow + 32 * i;
            const uint32_t l = (uint32_t)(uintptr_t)(sB + r * RS + cc4);
            const uint64_t g = (uint64_t)(uintptr_t)(W + (long)(n0 + r) * p.ldw + k0 + cc4);
            asm volatile("global_load_async_to_lds_b128 %0, %1, off"
                         :: "v"(l), "v"(g) : "memory");
        }
    };

    issue_stage(0, 0);
    asm volatile("s_wait_asynccnt 0x0" ::: "memory");
    __syncthreads();

    int stage = 0;
    for (int k0 = 0; k0 < p.K; k0 += BKT) {
        if (k0 + BKT < p.K) issue_stage(k0 + BKT, stage ^ 1);  // overlap w/ WMMA

        const float* sA = sm + stage * STAGEF;
        const float* sB = sA + BMT * RS;

        // A fragments: lane group0 K {0..7,16..23}, group1 K {8..15,24..31}
        const float* ar0 = sA + (wm * 32 + l16) * RS;
        const float* ar1 = sA + (wm * 32 + 16 + l16) * RS;
        v16bf a0 = cvt16(ar0 + 8 * half, ar0 + 16 + 8 * half);
        v16bf a1 = cvt16(ar1 + 8 * half, ar1 + 16 + 8 * half);

        // B fragments: lane group0 K {0..7,16..23}, group1 K {8..15,24..31}
        const float* br0 = sB + (wn * 32 + l16) * RS + 8 * half;
        const float* br1 = sB + (wn * 32 + 16 + l16) * RS + 8 * half;
        v16bf b0f = cvt16(br0, br0 + 16);
        v16bf b1f = cvt16(br1, br1 + 16);

        acc00 = __builtin_amdgcn_wmma_f32_16x16x32_bf16(
            false, a0, false, b0f, (short)0, acc00, false, false);
        acc01 = __builtin_amdgcn_wmma_f32_16x16x32_bf16(
            false, a0, false, b1f, (short)0, acc01, false, false);
        acc10 = __builtin_amdgcn_wmma_f32_16x16x32_bf16(
            false, a1, false, b0f, (short)0, acc10, false, false);
        acc11 = __builtin_amdgcn_wmma_f32_16x16x32_bf16(
            false, a1, false, b1f, (short)0, acc11, false, false);

        asm volatile("s_wait_asynccnt 0x0" ::: "memory");
        __syncthreads();
        stage ^= 1;
    }

    // C fragment layout: lane VGPR r -> (m = 8*(lane/16)+r, n = lane%16)
    #pragma unroll
    for (int r = 0; r < 8; ++r) {
        const int ml0 = m0 + wm * 32 + half * 8 + r;
        const int ml1 = ml0 + 16;
        const int nl0 = n0 + wn * 32 + l16;
        const int nl1 = nl0 + 16;
        if (!p.storeT) {
            const long i00 = (long)ml0 * p.ldc + nl0;
            const long i01 = (long)ml0 * p.ldc + nl1;
            const long i10 = (long)ml1 * p.ldc + nl0;
            const long i11 = (long)ml1 * p.ldc + nl1;
            float v00 = acc00[r], v01 = acc01[r], v10 = acc10[r], v11 = acc11[r];
            if (Add) { v00 += Add[i00]; v01 += Add[i01]; v10 += Add[i10]; v11 += Add[i11]; }
            C[i00] = v00; C[i01] = v01; C[i10] = v10; C[i11] = v11;
        } else {
            C[(long)nl0 * p.ldc + ml0] = acc00[r];
            C[(long)nl1 * p.ldc + ml0] = acc01[r];
            C[(long)nl0 * p.ldc + ml1] = acc10[r];
            C[(long)nl1 * p.ldc + ml1] = acc11[r];
        }
    }
}

// ---------------------------------------------------------------------------
// Helpers
// ---------------------------------------------------------------------------
__global__ void transpose_kernel(const float* __restrict__ src, float* __restrict__ dst,
                                 int R, int C, long sB, long dB) {
    const float* s = src + (long)blockIdx.z * sB;
    float*       d = dst + (long)blockIdx.z * dB;
    int i = blockIdx.x * blockDim.x + threadIdx.x;
    if (i < R * C) { int r = i / C, c = i % C; d[(long)c * R + r] = s[i]; }
}

__global__ void sub_kernel(const float* __restrict__ a, const float* __restrict__ b,
                           float* __restrict__ o, long n) {
    long i = (long)blockIdx.x * blockDim.x + threadIdx.x;
    if (i < n) o[i] = a[i] - b[i];
}

// one wave per row: softmax(scale * row)
__global__ void softmax_kernel(float* __restrict__ p, long rows, int cols, float scale) {
    long row = (long)blockIdx.x * (blockDim.x / 32) + (threadIdx.x >> 5);
    if (row >= rows) return;
    const int lane = threadIdx.x & 31;
    float* r = p + row * (long)cols;
    float mx = -INFINITY;
    for (int e = lane; e < cols; e += 32) mx = fmaxf(mx, r[e] * scale);
    for (int o = 16; o > 0; o >>= 1) mx = fmaxf(mx, __shfl_xor(mx, o, 32));
    float s = 0.f;
    for (int e = lane; e < cols; e += 32) { float v = __expf(r[e] * scale - mx); r[e] = v; s += v; }
    for (int o = 16; o > 0; o >>= 1) s += __shfl_xor(s, o, 32);
    const float inv = 1.0f / s;
    for (int e = lane; e < cols; e += 32) r[e] *= inv;
}

// one wave per token: iterative top-8 argmax (mutates coeffs)
__global__ void topk_kernel(float* __restrict__ coeffs, float* __restrict__ vals,
                            int* __restrict__ idx, int ntok, int dict) {
    int row = blockIdx.x * (blockDim.x / 32) + (threadIdx.x >> 5);
    if (row >= ntok) return;
    const int lane = threadIdx.x & 31;
    float* c = coeffs + (long)row * dict;
    for (int j = 0; j < TOP_K; ++j) {
        float best = -INFINITY; int bi = dict;
        for (int e = lane; e < dict; e += 32) {
            float v = c[e];
            if (v > best || (v == best && e < bi)) { best = v; bi = e; }
        }
        for (int o = 16; o > 0; o >>= 1) {
            float ov = __shfl_xor(best, o, 32);
            int   oi = __shfl_xor(bi,   o, 32);
            if (ov > best || (ov == best && oi < bi)) { best = ov; bi = oi; }
        }
        if (lane == (bi & 31)) c[bi] = -INFINITY;  // owner lane masks winner
        if (lane == 0) { vals[(long)row * TOP_K + j] = best; idx[(long)row * TOP_K + j] = bi; }
    }
}

// x_recon[t,:] = basis[t,:] + bias[:] + sum_j vals[t,j] * dict_w[idx[t,j],:]
__global__ void recon_kernel(float* __restrict__ xr, const float* __restrict__ vals,
                             const int* __restrict__ idx, const float* __restrict__ dict_w,
                             const float* __restrict__ bias) {
    const int t = blockIdx.x;
    __shared__ float sv[TOP_K]; __shared__ int si[TOP_K];
    if (threadIdx.x < TOP_K) {
        sv[threadIdx.x] = vals[(long)t * TOP_K + threadIdx.x];
        si[threadIdx.x] = idx[(long)t * TOP_K + threadIdx.x];
    }
    __syncthreads();
    for (int d = threadIdx.x; d < D_MODEL; d += blockDim.x) {
        float a = xr[(long)t * D_MODEL + d] + bias[d];
        #pragma unroll
        for (int j = 0; j < TOP_K; ++j) a += sv[j] * dict_w[(long)si[j] * D_MODEL + d];
        xr[(long)t * D_MODEL + d] = a;
    }
}

// per-token RMS norm; records rms for the e1/e2 means
__global__ void rmsnorm_kernel(const float* __restrict__ in, const float* __restrict__ w,
                               float* __restrict__ out, float* __restrict__ rms_arr) {
    const int t = blockIdx.x;
    __shared__ float red[8];
    float s = 0.f;
    for (int d = threadIdx.x; d < D_MODEL; d += blockDim.x) {
        float v = in[(long)t * D_MODEL + d]; s += v * v;
    }
    for (int o = 16; o > 0; o >>= 1) s += __shfl_xor(s, o, 32);
    if ((threadIdx.x & 31) == 0) red[threadIdx.x >> 5] = s;
    __syncthreads();
    if (threadIdx.x == 0) {
        float tt = 0.f;
        for (int i = 0; i < 8; ++i) tt += red[i];
        red[0] = sqrtf(tt / (float)D_MODEL + EPSV);
    }
    __syncthreads();
    const float rms = red[0];
    if (threadIdx.x == 0) rms_arr[t] = rms;
    const float inv = 1.0f / rms;
    for (int d = threadIdx.x; d < D_MODEL; d += blockDim.x)
        out[(long)t * D_MODEL + d] = w[d] * in[(long)t * D_MODEL + d] * inv;
}

// g = sigmoid(g) * silu(u)
__global__ void act_kernel(float* __restrict__ g, const float* __restrict__ u, long n) {
    long i = (long)blockIdx.x * blockDim.x + threadIdx.x;
    if (i < n) {
        float gv = g[i], uv = u[i];
        float sg = 1.0f / (1.0f + __expf(-gv));
        float su = uv / (1.0f + __expf(-uv));
        g[i] = sg * su;
    }
}

// deterministic single-block mean (optionally of |x|)
__global__ void reduce_mean_kernel(const float* __restrict__ src, int n,
                                   float* __restrict__ out, float scale, int useAbs) {
    __shared__ float red[8];
    float s = 0.f;
    for (int i = threadIdx.x; i < n; i += 256) {
        float v = src[i]; s += useAbs ? fabsf(v) : v;
    }
    for (int o = 16; o > 0; o >>= 1) s += __shfl_xor(s, o, 32);
    if ((threadIdx.x & 31) == 0) red[threadIdx.x >> 5] = s;
    __syncthreads();
    if (threadIdx.x == 0) {
        float t = 0.f;
        for (int i = 0; i < 8; ++i) t += red[i];
        out[0] = t * scale;
    }
}

// ---------------------------------------------------------------------------
// Host-side launch helper
// ---------------------------------------------------------------------------
static inline void launch_gemm(hipStream_t s, const float* A, const float* W, float* C,
                               const float* Add, int M, int N, int K,
                               int lda, int ldw, int ldc,
                               int nz = 1, int nb1 = 1,
                               long aS0 = 0, long aS1 = 0, long wS0 = 0, long wS1 = 0,
                               long cS0 = 0, long cS1 = 0, int storeT = 0) {
    GemmP p{A, W, C, Add, M, N, K, lda, ldw, ldc, nb1, storeT,
            aS0, aS1, wS0, wS1, cS0, cS1};
    dim3 g(N / BNT, M / BMT, nz);
    hipLaunchKernelGGL(gemm_bf16_wmma, g, dim3(256), 0, s, p);
}

extern "C" void kernel_launch(void* const* d_in, const int* in_sizes, int n_in,
                              void* d_out, int out_size, void* d_ws, size_t ws_size,
                              hipStream_t stream) {
    (void)in_sizes; (void)n_in; (void)out_size; (void)ws_size;
    const float* x         = (const float*)d_in[0];
    const float* proto     = (const float*)d_in[1];
    const float* gate_w    = (const float*)d_in[2];
    const float* dict_w    = (const float*)d_in[3];
    const float* enc_w     = (const float*)d_in[4];
    const float* wq        = (const float*)d_in[5];
    const float* wk        = (const float*)d_in[6];
    const float* wv        = (const float*)d_in[7];
    const float* wo        = (const float*)d_in[8];
    const float* rel_tmpl  = (const float*)d_in[9];
    const float* gate_proj = (const float*)d_in[10];
    const float* up_proj   = (const float*)d_in[11];
    const float* down_proj = (const float*)d_in[12];
    const float* norm1_w   = (const float*)d_in[13];
    const float* norm2_w   = (const float*)d_in[14];
    const float* bias_corr = (const float*)d_in[15];

    // --- d_out layout: x | attn | family_scores | {sparsity, e1, e2} ---
    float* out_x    = (float*)d_out;                         // 4096*1024
    float* out_attn = out_x + (long)NTOK * D_MODEL;          // 2*16*2048*2048
    float* out_fam  = out_attn + (long)BB * N_HEADS * TT * TT;
    float* out_scal = out_fam + (long)NTOK * NUM_FAM;        // 3 scalars

    // --- workspace layout (floats) ---
    float* ws = (float*)d_ws;
    long o = 0;
    float* protoT = ws + o; o += (long)D_MODEL * NUM_FAM;          // [1024,64]
    float* relT   = ws + o; o += (long)N_HEADS * D_HEAD * D_HEAD;  // [16][64,64] (e,d)
    float* vals   = ws + o; o += (long)NTOK * TOP_K;
    int*   idxb   = (int*)(ws + o); o += (long)NTOK * TOP_K;
    float* rms1   = ws + o; o += NTOK;
    float* rms2   = ws + o; o += NTOK;
    float* xr     = ws + o; o += (long)NTOK * D_MODEL;  // basis -> x_recon
    float* bufA   = ws + o; o += (long)NTOK * D_MODEL;  // resid, later attn-out
    float* bufB   = ws + o; o += (long)NTOK * D_MODEL;  // normed1, later normed2
    float* qb     = ws + o; o += (long)NTOK * D_MODEL;
    float* kb     = ws + o; o += (long)NTOK * D_MODEL;
    float* vTb    = ws + o; o += (long)NTOK * D_MODEL;  // [1024, 4096] transposed V
    float* kmodb  = ws + o; o += (long)NTOK * D_MODEL;  // later x2 (post-attn resid)
    float* big0   = ws + o; o += (long)NTOK * DICTN;    // coeffs, later gate
    float* big1   = ws + o; o += (long)NTOK * D_FF;     // up

    const long TD  = (long)TT * D_MODEL;      // per-batch token stride
    const long HTT = (long)N_HEADS * TT * TT; // per-batch attn stride

    // 0) transposes: proto [64,1024] -> protoT [1024,64]; rel_templates per head
    hipLaunchKernelGGL(transpose_kernel, dim3((NUM_FAM * D_MODEL + 255) / 256, 1, 1),
                       dim3(256), 0, stream, proto, protoT, NUM_FAM, D_MODEL, 0L, 0L);
    hipLaunchKernelGGL(transpose_kernel, dim3((D_HEAD * D_HEAD + 255) / 256, 1, N_HEADS),
                       dim3(256), 0, stream, rel_tmpl, relT, D_HEAD, D_HEAD,
                       (long)D_HEAD * D_HEAD, (long)D_HEAD * D_HEAD);

    // 1) family scores = softmax(x @ gate_w^T)   [4096,64]
    launch_gemm(stream, x, gate_w, out_fam, nullptr, NTOK, NUM_FAM, D_MODEL,
                D_MODEL, D_MODEL, NUM_FAM);
    hipLaunchKernelGGL(softmax_kernel, dim3(NTOK / 8), dim3(256), 0, stream,
                       out_fam, (long)NTOK, NUM_FAM, 1.0f);

    // 2) basis = fam @ proto  ==  fam @ protoT^T   -> xr
    launch_gemm(stream, out_fam, protoT, xr, nullptr, NTOK, D_MODEL, NUM_FAM,
                NUM_FAM, NUM_FAM, D_MODEL);

    // 3) resid = x - basis
    hipLaunchKernelGGL(sub_kernel, dim3((NTOK * D_MODEL) / 256), dim3(256), 0, stream,
                       x, xr, bufA, (long)NTOK * D_MODEL);

    // 4) coeffs = resid @ enc_w^T   [4096,4096]
    launch_gemm(stream, bufA, enc_w, big0, nullptr, NTOK, DICTN, D_MODEL,
                D_MODEL, D_MODEL, DICTN);

    // 5) top-8 per token (mutates big0)
    hipLaunchKernelGGL(topk_kernel, dim3(NTOK / 8), dim3(256), 0, stream,
                       big0, vals, idxb, NTOK, DICTN);

    // 6) x_recon = basis + offset + bias_corr (in place on xr)
    hipLaunchKernelGGL(recon_kernel, dim3(NTOK), dim3(256), 0, stream,
                       xr, vals, idxb, dict_w, bias_corr);

    // 7) normed1 = hcn(x_recon) -> bufB; rms -> rms1
    hipLaunchKernelGGL(rmsnorm_kernel, dim3(NTOK), dim3(256), 0, stream,
                       xr, norm1_w, bufB, rms1);

    // 8) q/k projections; v stored transposed [1024, 4096]
    launch_gemm(stream, bufB, wq, qb, nullptr, NTOK, D_MODEL, D_MODEL,
                D_MODEL, D_MODEL, D_MODEL);
    launch_gemm(stream, bufB, wk, kb, nullptr, NTOK, D_MODEL, D_MODEL,
                D_MODEL, D_MODEL, D_MODEL);
    launch_gemm(stream, bufB, wv, vTb, nullptr, NTOK, D_MODEL, D_MODEL,
                D_MODEL, D_MODEL, NTOK, /*nz*/1, 1, 0,0,0,0,0,0, /*storeT*/1);

    // 9) k_mod[b,h] = k[b,h] @ R_h  (W = relT_h, batched over (b,h))
    launch_gemm(stream, kb, relT, kmodb, nullptr, TT, D_HEAD, D_HEAD,
                D_MODEL, D_HEAD, D_MODEL,
                /*nz*/BB * N_HEADS, /*nb1*/N_HEADS,
                TD, D_HEAD, 0, (long)D_HEAD * D_HEAD, TD, D_HEAD);

    // 10) scores[b,h] = q @ k_mod^T -> directly into attn output slice
    launch_gemm(stream, qb, kmodb, out_attn, nullptr, TT, TT, D_HEAD,
                D_MODEL, D_MODEL, TT,
                BB * N_HEADS, N_HEADS,
                TD, D_HEAD, TD, D_HEAD, HTT, (long)TT * TT);

    // 11) softmax rows with 1/sqrt(64) scaling, in place
    hipLaunchKernelGGL(softmax_kernel, dim3((BB * N_HEADS * TT) / 8), dim3(256), 0, stream,
                       out_attn, (long)BB * N_HEADS * TT, TT, 0.125f);

    // 12) ctx[b,h] = attn @ v  (W = vT slice [64 rows, ldw=4096]) -> bufA [token,1024]
    launch_gemm(stream, out_attn, vTb, bufA, nullptr, TT, D_HEAD, TT,
                TT, NTOK, D_MODEL,
                BB * N_HEADS, N_HEADS,
                HTT, (long)TT * TT, (long)TT, (long)D_HEAD * NTOK, TD, D_HEAD);

    // 13) x2 = x + ctx @ wo^T   -> kmodb (reused)
    launch_gemm(stream, bufA, wo, kmodb, x, NTOK, D_MODEL, D_MODEL,
                D_MODEL, D_MODEL, D_MODEL);

    // 14) normed2 = hcn(x2) -> bufB; rms -> rms2
    hipLaunchKernelGGL(rmsnorm_kernel, dim3(NTOK), dim3(256), 0, stream,
                       kmodb, norm2_w, bufB, rms2);

    // 15) MLP: gate/up
    launch_gemm(stream, bufB, gate_proj, big0, nullptr, NTOK, D_FF, D_MODEL,
                D_MODEL, D_MODEL, D_FF);
    launch_gemm(stream, bufB, up_proj, big1, nullptr, NTOK, D_FF, D_MODEL,
                D_MODEL, D_MODEL, D_FF);
    hipLaunchKernelGGL(act_kernel, dim3(((long)NTOK * D_FF) / 256), dim3(256), 0, stream,
                       big0, big1, (long)NTOK * D_FF);

    // 16) x_out = x2 + act @ down_proj^T  -> d_out
    launch_gemm(stream, big0, down_proj, out_x, kmodb, NTOK, D_MODEL, D_FF,
                D_FF, D_FF, D_MODEL);

    // 17) scalars: sparsity = sum|vals| / (B*T*DICT); e1/e2 = mean(rms)
    hipLaunchKernelGGL(reduce_mean_kernel, dim3(1), dim3(256), 0, stream,
                       vals, NTOK * TOP_K, out_scal + 0,
                       1.0f / ((float)NTOK * (float)DICTN), 1);
    hipLaunchKernelGGL(reduce_mean_kernel, dim3(1), dim3(256), 0, stream,
                       rms1, NTOK, out_scal + 1, 1.0f / (float)NTOK, 0);
    hipLaunchKernelGGL(reduce_mean_kernel, dim3(1), dim3(256), 0, stream,
                       rms2, NTOK, out_scal + 2, 1.0f / (float)NTOK, 0);
}
